// QuantizedLinear_40647570489919
// MI455X (gfx1250) — matrix operand order repaired and checked by reading, not compiled, not run on hardware
//
#include <hip/hip_runtime.h>

typedef __attribute__((ext_vector_type(16))) _Float16 v16h;
typedef __attribute__((ext_vector_type(8)))  _Float16 v8h;
typedef __attribute__((ext_vector_type(8)))  float    v8f;
typedef __attribute__((ext_vector_type(8)))  int      v8i;
typedef __attribute__((ext_vector_type(4)))  int      v4i;

static constexpr int M_DIM = 8192;    // B * S
static constexpr int N_DIM = 11008;   // D_OUT
static constexpr int K_DIM = 4096;    // D_IN

// ---------------- CDNA5 async-copy plumbing ----------------

#if defined(__has_builtin)
#if __has_builtin(__builtin_amdgcn_global_load_async_to_lds_b128)
#define ASYNC_LDS 1
#endif
#endif

#if defined(ASYNC_LDS)
typedef __attribute__((address_space(1))) v4i gv4i;   // global
typedef __attribute__((address_space(3))) v4i lv4i;   // LDS
#endif

__device__ __forceinline__ void copy16(const _Float16* g, _Float16* l) {
#if defined(ASYNC_LDS)
    __builtin_amdgcn_global_load_async_to_lds_b128((gv4i*)g, (lv4i*)l, 0, 0);
#else
    v4i t = *(const v4i*)g;
    *(v4i*)l = t;
#endif
}

__device__ __forceinline__ void wait_async_all() {
#if defined(ASYNC_LDS)
#if __has_builtin(__builtin_amdgcn_s_wait_asynccnt)
    __builtin_amdgcn_s_wait_asynccnt(0);
#else
    asm volatile("s_wait_asynccnt 0x0" ::: "memory");
#endif
#endif
}

// ---------------- pre-conversion kernels (memory-bound, one pass) ----------------

__global__ void cvt_x_f16(const float* __restrict__ in, _Float16* __restrict__ out, long n8) {
    long i = (long)blockIdx.x * blockDim.x + threadIdx.x;
    if (i >= n8) return;
    long base = i * 8;
    v8f v = *(const v8f*)(in + base);
    v8h h;
#pragma unroll
    for (int j = 0; j < 8; ++j) h[j] = (_Float16)v[j];
    *(v8h*)(out + base) = h;
}

__global__ void cvt_w_f16(const int* __restrict__ in, _Float16* __restrict__ out, long n8) {
    long i = (long)blockIdx.x * blockDim.x + threadIdx.x;
    if (i >= n8) return;
    long base = i * 8;
    v8i v = *(const v8i*)(in + base);
    v8h h;
#pragma unroll
    for (int j = 0; j < 8; ++j) h[j] = (_Float16)(float)v[j];   // codes 0..127: exact in f16
    *(v8h*)(out + base) = h;
}

// ---------------- main GEMM: async double-buffered LDS pipeline ----------------
// Workgroup 128(M) x 128(N); 8 waves as 2(M) x 4(N); wave tile 64x32 = 4x2 WMMA accums.
// K-block 64, two LDS stages; 72-half row stride for bank-conflict-free ds_load_b128.

static constexpr int BK      = 64;
static constexpr int LDR     = BK + 8;        // 72 halfs = 144 B row stride
static constexpr int STAGE_H = 128 * LDR;     // halfs per matrix per stage

union AFrag { v16h v; v8h h[2]; };

__global__ __launch_bounds__(256)
void qlinear_wmma_lds(const _Float16* __restrict__ Ah, const _Float16* __restrict__ Wh,
                      const float* __restrict__ scale, const float* __restrict__ bias,
                      float* __restrict__ out)
{
    extern __shared__ _Float16 smem[];
    _Float16* sA = smem;                    // [2][STAGE_H]
    _Float16* sB = smem + 2 * STAGE_H;      // [2][STAGE_H]

    const int tid  = threadIdx.x;
    const int lane = tid & 31;
    const int wid  = tid >> 5;
    const int wm   = wid >> 2;              // 0..1
    const int wn   = wid & 3;               // 0..3
    const int mBlk = blockIdx.y * 128;
    const int nBlk = blockIdx.x * 128;

    const int sub = lane >> 4;              // lane half: selects K sub-range
    const int rA  = lane & 15;              // A row-in-tile / B,C column-in-tile

    // global->LDS copy mapping: 128 rows x 8 chunks (16B) per matrix per stage,
    // 1024 chunks / 256 threads = 4 per thread, fully coalesced.
    const int lrow = tid >> 3;              // 0..31
    const int lch  = (tid & 7) * 8;         // half offset of 16B chunk in row

    v8f acc[4][2];
#pragma unroll
    for (int t = 0; t < 4; ++t) {
#pragma unroll
        for (int u = 0; u < 2; ++u) acc[t][u] = {};
    }

    auto issue = [&](int kb, int stg) {
        const _Float16* gA = Ah + (long)mBlk * K_DIM + kb;
        const _Float16* gB = Wh + (long)nBlk * K_DIM + kb;
        _Float16* dA = sA + stg * STAGE_H;
        _Float16* dB = sB + stg * STAGE_H;
#pragma unroll
        for (int j = 0; j < 4; ++j) {
            const int r = lrow + 32 * j;
            copy16(gA + (long)r * K_DIM + lch, dA + r * LDR + lch);
            copy16(gB + (long)r * K_DIM + lch, dB + r * LDR + lch);
        }
    };

    issue(0, 0);
    wait_async_all();
    __syncthreads();

    const int nSteps = K_DIM / BK;          // 64
    for (int s = 0; s < nSteps; ++s) {
        const int stg = s & 1;
        if (s + 1 < nSteps) issue((s + 1) * BK, stg ^ 1);

        const _Float16* tA = sA + stg * STAGE_H;
        const _Float16* tB = sB + stg * STAGE_H;
#pragma unroll
        for (int ks = 0; ks < BK; ks += 32) {
            v16h aF[4];
#pragma unroll
            for (int t = 0; t < 4; ++t) {
                const int row = wm * 64 + t * 16 + rA;
                const _Float16* p = tA + row * LDR + ks + sub * 8;
                AFrag a;
                a.h[0] = *(const v8h*)(p);
                a.h[1] = *(const v8h*)(p + 16);
                aF[t] = a.v;
            }
            v16h bF[2];
#pragma unroll
            for (int u = 0; u < 2; ++u) {
                const int row = wn * 32 + u * 16 + rA;
                const _Float16* p = tB + row * LDR + ks + sub * 8;
                AFrag b;
                b.h[0] = *(const v8h*)(p);
                b.h[1] = *(const v8h*)(p + 16);
                bF[u] = b.v;
            }
#pragma unroll
            for (int t = 0; t < 4; ++t) {
#pragma unroll
                for (int u = 0; u < 2; ++u) {
                    acc[t][u] = __builtin_amdgcn_wmma_f32_16x16x32_f16(
                        false, aF[t], false, bF[u], (short)0, acc[t][u], false, false);
                }
            }
        }
        if (s + 1 < nSteps) wait_async_all();
        __syncthreads();
    }

    // ---- epilogue: apply per-tensor scale + bias, store f32 ----
    const float sc = scale[0];
    const int m0 = mBlk + wm * 64;
    const int n0 = nBlk + wn * 32;
    const int rowBase = sub * 8;
#pragma unroll
    for (int u = 0; u < 2; ++u) {
        const int col = n0 + u * 16 + rA;
        const float bv = bias[col];
#pragma unroll
        for (int t = 0; t < 4; ++t) {
#pragma unroll
            for (int r = 0; r < 8; ++r) {
                const int row = m0 + t * 16 + rowBase + r;
                out[(long)row * N_DIM + col] = acc[t][u][r] * sc + bv;
            }
        }
    }
}

// ---------------- fallback: fused-conversion direct-global kernel ----------------

__global__ __launch_bounds__(256)
void qlinear_wmma_direct(const float* __restrict__ Af, const int* __restrict__ Wi,
                         const float* __restrict__ scale, const float* __restrict__ bias,
                         float* __restrict__ out)
{
    const int lane = threadIdx.x & 31;
    const int wid  = threadIdx.x >> 5;
    const int wm   = wid >> 2;
    const int wn   = wid & 3;
    const int m0   = blockIdx.y * 128 + wm * 64;
    const int n0   = blockIdx.x * 128 + wn * 32;
    const int sub  = lane >> 4;
    const int rA   = lane & 15;

    v8f acc[4][2];
#pragma unroll
    for (int t = 0; t < 4; ++t) {
#pragma unroll
        for (int u = 0; u < 2; ++u) acc[t][u] = {};
    }

    for (int kb = 0; kb < K_DIM; kb += 32) {
        v16h aF[4];
#pragma unroll
        for (int t = 0; t < 4; ++t) {
            const long off = (long)(m0 + t * 16 + rA) * K_DIM + kb + sub * 8;
            v8f f0 = *(const v8f*)(Af + off);
            v8f f1 = *(const v8f*)(Af + off + 16);
            AFrag a;
#pragma unroll
            for (int j = 0; j < 8; ++j) {
                a.h[0][j] = (_Float16)f0[j];
                a.h[1][j] = (_Float16)f1[j];
            }
            aF[t] = a.v;
        }
        v16h bF[2];
#pragma unroll
        for (int u = 0; u < 2; ++u) {
            const long off = (long)(n0 + u * 16 + rA) * K_DIM + kb + sub * 8;
            v8i w0 = *(const v8i*)(Wi + off);
            v8i w1 = *(const v8i*)(Wi + off + 16);
            v16h b;
#pragma unroll
            for (int j = 0; j < 8; ++j) {
                b[j]     = (_Float16)(float)w0[j];
                b[8 + j] = (_Float16)(float)w1[j];
            }
            bF[u] = b;
        }
#pragma unroll
        for (int t = 0; t < 4; ++t) {
#pragma unroll
            for (int u = 0; u < 2; ++u) {
                acc[t][u] = __builtin_amdgcn_wmma_f32_16x16x32_f16(
                    false, aF[t], false, bF[u], (short)0, acc[t][u], false, false);
            }
        }
    }

    const float sc = scale[0];
    const int rowBase = sub * 8;
#pragma unroll
    for (int u = 0; u < 2; ++u) {
        const int col = n0 + u * 16 + rA;
        const float bv = bias[col];
#pragma unroll
        for (int t = 0; t < 4; ++t) {
#pragma unroll
            for (int r = 0; r < 8; ++r) {
                const int row = m0 + t * 16 + rowBase + r;
                out[(long)row * N_DIM + col] = acc[t][u][r] * sc + bv;
            }
        }
    }
}

// ---------------- host launch ----------------

extern "C" void kernel_launch(void* const* d_in, const int* in_sizes, int n_in,
                              void* d_out, int out_size, void* d_ws, size_t ws_size,
                              hipStream_t stream) {
    const float* x     = (const float*)d_in[0];
    const int*   wq    = (const int*)d_in[1];
    const float* scale = (const float*)d_in[2];
    const float* bias  = (const float*)d_in[3];
    float*       out   = (float*)d_out;

    const size_t nA = (size_t)M_DIM * K_DIM;
    const size_t nW = (size_t)N_DIM * K_DIM;
    const size_t bytesA = nA * 2;               // f16
    const size_t bytesW = nW * 2;               // f16

    dim3 grid(N_DIM / 128, M_DIM / 128);        // 86 x 64 workgroups
    dim3 block(256);

    if (ws_size >= bytesA + bytesW) {
        _Float16* Ahp = (_Float16*)d_ws;
        _Float16* Whp = (_Float16*)((char*)d_ws + bytesA);
        const long n8A = (long)(nA / 8);
        const long n8W = (long)(nW / 8);
        cvt_x_f16<<<dim3((unsigned)((n8A + 255) / 256)), dim3(256), 0, stream>>>(x, Ahp, n8A);
        cvt_w_f16<<<dim3((unsigned)((n8W + 255) / 256)), dim3(256), 0, stream>>>(wq, Whp, n8W);
        const size_t ldsBytes = (size_t)(4 * STAGE_H) * 2;   // 73,728 B
        qlinear_wmma_lds<<<grid, block, ldsBytes, stream>>>(Ahp, Whp, scale, bias, out);
    } else {
        qlinear_wmma_direct<<<grid, block, 0, stream>>>(x, wq, scale, bias, out);
    }
}
